// HierarchicalSparseAttention_62534723830440
// MI455X (gfx1250) — matrix operand order repaired and checked by reading, not compiled, not run on hardware
//
#include <hip/hip_runtime.h>
#include <hip/hip_bf16.h>

typedef __attribute__((ext_vector_type(16))) __bf16 v16bf;
typedef __attribute__((ext_vector_type(8)))  float   v8f;
typedef int v4i __attribute__((vector_size(16)));

#define SEQ_N   2048
#define DIM_D   1024
#define HEADS   16
#define DH      64
#define KV_ROWS 4095   // 2N-1
#define WINDOW  22
#define NKEYS   54     // (2*22-1) + log2(2048)

#if defined(__AMDGCN__) && __has_builtin(__builtin_amdgcn_global_load_async_to_lds_b128) && \
    __has_builtin(__builtin_amdgcn_s_wait_asynccnt)
#define HAVE_ASYNC_LDS 1
#else
#define HAVE_ASYNC_LDS 0
#endif

// stage one 16-byte chunk global -> LDS
__device__ __forceinline__ void stage16(const __bf16* g, __bf16* l) {
#if HAVE_ASYNC_LDS
  __builtin_amdgcn_global_load_async_to_lds_b128(
      (__attribute__((address_space(1))) v4i*)g,
      (__attribute__((address_space(3))) v4i*)l, 0, 0);
#else
  *(uint4*)l = *(const uint4*)g;
#endif
}

// ---------------------------------------------------------------------------
// fp32 -> bf16 conversion (grid-stride)
// ---------------------------------------------------------------------------
__global__ void cvt_f32_to_bf16(const float* __restrict__ in,
                                __bf16* __restrict__ out, int n) {
  int i = blockIdx.x * blockDim.x + threadIdx.x;
  int stride = gridDim.x * blockDim.x;
  for (; i < n; i += stride) out[i] = (__bf16)in[i];
}

// ---------------------------------------------------------------------------
// Tiled bf16 WMMA GEMM:  C[M,Nn] = A[M,K] * Bt[Nn,K]^T (+bias)
// Block tile 128x128, BK=64, 8 waves; wave tile 32x64 (2x4 of 16x16).
// 16 WMMAs per barrier period, LDS staged via async-to-LDS when available.
// Cb!=null -> store bf16, else store fp32 (+optional bias).
// ---------------------------------------------------------------------------
__global__ __launch_bounds__(256)
void gemm_bf16_wmma(const __bf16* __restrict__ A, const __bf16* __restrict__ Bt,
                    float* __restrict__ Cf, __bf16* __restrict__ Cb,
                    const float* __restrict__ bias, int M, int Nn, int K) {
  constexpr int BM = 128, BN = 128, BK = 64;
  constexpr int LDH = 72;               // halves per LDS row (64 data + 8 pad) = 144B
  __shared__ alignas(16) __bf16 As[BM * LDH];
  __shared__ alignas(16) __bf16 Bs[BN * LDH];

  const int tid = threadIdx.x;
  const int wave = tid >> 5, lane = tid & 31;
  const int wm = (wave & 3) * 32;       // wave M offset in block tile (4 waves)
  const int wn = (wave >> 2) * 64;      // wave N offset in block tile (2 waves)
  const int m0 = blockIdx.x * BM;
  const int n0 = blockIdx.y * BN;
  const int laneRow = lane & 15;
  const int laneHi = lane >> 4;         // 0 or 1

  v8f acc[2][4] = {};

  // staging: each thread covers one half-row (32 halves = 4 x b128)
  const int srow = tid >> 1;                 // 0..127
  const int scol = (tid & 1) * 32;           // 0 or 32 halves

  int garow = m0 + srow; if (garow >= M) garow = M - 1;   // clamp (stores guarded)
  const __bf16* gA = A + (size_t)garow * K + scol;
  const __bf16* gB = Bt + (size_t)(n0 + srow) * K + scol;
  __bf16* lA = &As[srow * LDH + scol];
  __bf16* lB = &Bs[srow * LDH + scol];

  for (int k0 = 0; k0 < K; k0 += BK) {
#pragma unroll
    for (int c = 0; c < 32; c += 8) {
      stage16(gA + k0 + c, lA + c);
      stage16(gB + k0 + c, lB + c);
    }
#if HAVE_ASYNC_LDS
    __builtin_amdgcn_s_wait_asynccnt(0);
#endif
    __syncthreads();

    if (k0 + BK < K) {                  // prefetch next K tile
      __builtin_prefetch(gA + k0 + BK, 0, 3);
      __builtin_prefetch(gB + k0 + BK, 0, 3);
    }

#pragma unroll
    for (int ks = 0; ks < BK; ks += 32) {
      // A fragment: lane holds row m, k = {ks+laneHi*8+0..7, ks+16+laneHi*8+0..7}
      v16bf afrag[2], bfrag[4];
#pragma unroll
      for (int mi = 0; mi < 2; ++mi) {
        const __bf16* p = &As[(wm + mi * 16 + laneRow) * LDH + ks + laneHi * 8];
        ((uint4*)&afrag[mi])[0] = *(const uint4*)(p);
        ((uint4*)&afrag[mi])[1] = *(const uint4*)(p + 16);
      }
      // B fragment: lane holds col n (= row of Bt), k = {ks+laneHi*8+0..7, ks+16+laneHi*8+0..7}
#pragma unroll
      for (int ni = 0; ni < 4; ++ni) {
        const __bf16* p = &Bs[(wn + ni * 16 + laneRow) * LDH + ks + laneHi * 8];
        ((uint4*)&bfrag[ni])[0] = *(const uint4*)(p);
        ((uint4*)&bfrag[ni])[1] = *(const uint4*)(p + 16);
      }
#pragma unroll
      for (int mi = 0; mi < 2; ++mi)
#pragma unroll
        for (int ni = 0; ni < 4; ++ni)
          acc[mi][ni] = __builtin_amdgcn_wmma_f32_16x16x32_bf16(
              false, afrag[mi], false, bfrag[ni], (short)0, acc[mi][ni],
              false, false);
    }
    __syncthreads();
  }

  // epilogue: VGPR r -> row r + laneHi*8, col = laneRow
#pragma unroll
  for (int mi = 0; mi < 2; ++mi) {
#pragma unroll
    for (int ni = 0; ni < 4; ++ni) {
      const int gn = n0 + wn + ni * 16 + laneRow;
#pragma unroll
      for (int r = 0; r < 8; ++r) {
        const int gm = m0 + wm + mi * 16 + laneHi * 8 + r;
        if (gm < M) {
          float v = acc[mi][ni][r];
          if (Cb) Cb[(size_t)gm * Nn + gn] = (__bf16)v;
          else    Cf[(size_t)gm * Nn + gn] = v + (bias ? bias[gn] : 0.0f);
        }
      }
    }
  }
}

// ---------------------------------------------------------------------------
// Sparse attention: one wave per (query l, head h); lanes span the 64-dim head
// (d = lane, lane+32). Online softmax over 43 window keys + 11 hierarchical.
// ---------------------------------------------------------------------------
__global__ __launch_bounds__(256)
void attn_kernel(const __bf16* __restrict__ Qb, const __bf16* __restrict__ Kb,
                 const __bf16* __restrict__ Vb, __bf16* __restrict__ Ob,
                 const int* __restrict__ maskp) {
  const int wave = threadIdx.x >> 5;
  const int lane = threadIdx.x & 31;
  const int qh = blockIdx.x * 8 + wave;
  const int h = qh & (HEADS - 1);
  const int l = qh >> 4;
  const bool causal = (maskp[0] != 0);
  const float scale = 0.125f;  // 1/sqrt(64)

  const size_t qoff = (size_t)l * DIM_D + h * DH;
  const float q0 = (float)Qb[qoff + lane];
  const float q1 = (float)Qb[qoff + lane + 32];

  float m = -INFINITY, s = 0.0f, o0 = 0.0f, o1 = 0.0f;

  for (int j = 0; j < NKEYS; ++j) {
    int idx; bool ok;
    if (j < 2 * WINDOW - 1) {                   // sliding window
      idx = l - (WINDOW - 1) + j;
      ok = (idx >= 0) && (idx < SEQ_N) && (!causal || idx <= l);
    } else {                                    // hierarchical chain
      const int lvl_t = j - (2 * WINDOW - 1);   // 0..10
      int nc = l; bool broke = false; bool cm = false;
      for (int lvl = 0; lvl <= lvl_t; ++lvl) {
        int nxt, pair;
        if (lvl == 0) { nxt = nc ^ 1; pair = nc; }
        else          { pair = (nc >> 1) + SEQ_N; nxt = pair ^ 1; }
        if (nxt > 2 * SEQ_N - 3) { broke = true; break; }  // > max_valid
        cm = (pair < nxt);
        nc = nxt;
      }
      if (broke) { idx = 2 * SEQ_N - 2; ok = true; }   // -1 wraps to 4094, cm=False
      else       { idx = nc; ok = !(causal && cm); }
    }
    if (ok) {
      const size_t koff = (size_t)idx * DIM_D + h * DH;
      const float k0 = (float)Kb[koff + lane];
      const float k1 = (float)Kb[koff + lane + 32];
      float p = q0 * k0 + q1 * k1;
#pragma unroll
      for (int off = 16; off > 0; off >>= 1) p += __shfl_xor(p, off, 32);
      const float logit = p * scale;
      const float mnew = fmaxf(m, logit);
      const float c = __expf(m - mnew);
      const float w = __expf(logit - mnew);
      const float v0 = (float)Vb[koff + lane];
      const float v1 = (float)Vb[koff + lane + 32];
      s  = s  * c + w;
      o0 = o0 * c + w * v0;
      o1 = o1 * c + w * v1;
      m = mnew;
    }
  }
  const float inv = 1.0f / s;
  Ob[qoff + lane]      = (__bf16)(o0 * inv);
  Ob[qoff + lane + 32] = (__bf16)(o1 * inv);
}

// ---------------------------------------------------------------------------
extern "C" void kernel_launch(void* const* d_in, const int* in_sizes, int n_in,
                              void* d_out, int out_size, void* d_ws, size_t ws_size,
                              hipStream_t stream) {
  (void)in_sizes; (void)n_in; (void)out_size; (void)ws_size;
  const float* x   = (const float*)d_in[0];   // query (== x)
  // d_in[1] (key) and d_in[2] (value) are unused by the reference
  const float* y   = (const float*)d_in[3];
  const float* Wq  = (const float*)d_in[4];
  const float* Wk  = (const float*)d_in[5];
  const float* Wv  = (const float*)d_in[6];
  const float* Wo  = (const float*)d_in[7];
  const float* Wob = (const float*)d_in[8];
  const int*   msk = (const int*)d_in[9];

  size_t off = 0;
  auto alloc = [&](size_t bytes) -> void* {
    void* p = (char*)d_ws + off;
    off += (bytes + 255) & ~(size_t)255;
    return p;
  };
  __bf16* XYb = (__bf16*)alloc((size_t)KV_ROWS * DIM_D * 2);
  __bf16* Wqb = (__bf16*)alloc((size_t)DIM_D * DIM_D * 2);
  __bf16* Wkb = (__bf16*)alloc((size_t)DIM_D * DIM_D * 2);
  __bf16* Wvb = (__bf16*)alloc((size_t)DIM_D * DIM_D * 2);
  __bf16* Wob_h = (__bf16*)alloc((size_t)DIM_D * DIM_D * 2);
  __bf16* Qb  = (__bf16*)alloc((size_t)SEQ_N * DIM_D * 2);
  __bf16* Kb  = (__bf16*)alloc((size_t)KV_ROWS * DIM_D * 2);
  __bf16* Vb  = (__bf16*)alloc((size_t)KV_ROWS * DIM_D * 2);
  __bf16* Ab  = (__bf16*)alloc((size_t)SEQ_N * DIM_D * 2);

  const int CVB = 2048;
  cvt_f32_to_bf16<<<CVB, 256, 0, stream>>>(x, XYb, SEQ_N * DIM_D);
  cvt_f32_to_bf16<<<CVB, 256, 0, stream>>>(y, XYb + (size_t)SEQ_N * DIM_D,
                                           (SEQ_N - 1) * DIM_D);
  cvt_f32_to_bf16<<<CVB, 256, 0, stream>>>(Wq, Wqb, DIM_D * DIM_D);
  cvt_f32_to_bf16<<<CVB, 256, 0, stream>>>(Wk, Wkb, DIM_D * DIM_D);
  cvt_f32_to_bf16<<<CVB, 256, 0, stream>>>(Wv, Wvb, DIM_D * DIM_D);
  cvt_f32_to_bf16<<<CVB, 256, 0, stream>>>(Wo, Wob_h, DIM_D * DIM_D);

  // Q = x @ Wq^T  (first 2048 rows of XYb)
  gemm_bf16_wmma<<<dim3(SEQ_N / 128, DIM_D / 128), 256, 0, stream>>>(
      XYb, Wqb, nullptr, Qb, nullptr, SEQ_N, DIM_D, DIM_D);
  // K = XY @ Wk^T ; V = XY @ Wv^T  (4095 rows)
  gemm_bf16_wmma<<<dim3((KV_ROWS + 127) / 128, DIM_D / 128), 256, 0, stream>>>(
      XYb, Wkb, nullptr, Kb, nullptr, KV_ROWS, DIM_D, DIM_D);
  gemm_bf16_wmma<<<dim3((KV_ROWS + 127) / 128, DIM_D / 128), 256, 0, stream>>>(
      XYb, Wvb, nullptr, Vb, nullptr, KV_ROWS, DIM_D, DIM_D);

  // sparse attention -> bf16 activations
  attn_kernel<<<(SEQ_N * HEADS) / 8, 256, 0, stream>>>(Qb, Kb, Vb, Ab, msk);

  // out = attn @ Wo^T + b   (fp32 to d_out)
  gemm_bf16_wmma<<<dim3(SEQ_N / 128, DIM_D / 128), 256, 0, stream>>>(
      Ab, Wob_h, (float*)d_out, nullptr, Wob, SEQ_N, DIM_D, DIM_D);
}
